// RNN_50044958933139
// MI455X (gfx1250) — matrix operand order repaired and checked by reading, not compiled, not run on hardware
//
#include <hip/hip_runtime.h>
#include <hip/hip_bf16.h>

typedef __attribute__((ext_vector_type(16))) __bf16 v16bf;
typedef __attribute__((ext_vector_type(8)))  float  v8f;

#define WMMA_BF16(a, b, c) \
  __builtin_amdgcn_wmma_f32_16x16x32_bf16(false, (a), false, (b), (short)0, (c), false, false)

// Sizes
#define SEQ   128
#define BATCH 128
#define VOCAB 10000
#define EMB   512
#define HID   512

// A-fragment (16-bit A 16x32, ISA 7.12.2): lane -> M = lane&15, lane>>4 selects K half.
__device__ __forceinline__ int afrag_k(int lane, int idx) {
  int khalf = lane >> 4;
  int r = idx >> 1, c = idx & 1;
  return (r < 4) ? (khalf * 8 + r * 2 + c) : (16 + khalf * 8 + (r - 4) * 2 + c);
}
// B-fragment (32x16): lane -> N = lane&15, K = (lane>>4)*8 + idx for idx<8, else 16 + (lane>>4)*8 + (idx-8)

// ---------------------------------------------------------------------------
// Prep: swizzle weight matrix W[N,K] (f32, row major) into bf16 B-fragment order
// dst layout: [ntile][ktile][lane*16+idx], flat == tid
// ---------------------------------------------------------------------------
__global__ void prep_w(const float* __restrict__ W, __bf16* __restrict__ dst,
                       int total, int ktbits, int K) {
  int tid = blockIdx.x * 256 + threadIdx.x;
  if (tid >= total) return;
  int idx  = tid & 15;
  int lane = (tid >> 4) & 31;
  int kt   = (tid >> 9) & ((1 << ktbits) - 1);
  int nt   = tid >> (9 + ktbits);
  int n = nt * 16 + (lane & 15);
  int k = kt * 32 + ((idx < 8) ? ((lane >> 4) * 8 + idx) : (16 + (lane >> 4) * 8 + idx - 8));
  dst[tid] = (__bf16)W[(size_t)n * K + k];
}

// ---------------------------------------------------------------------------
// Prep: gather embeddings for every (s, b) and store in A-fragment order
// dst layout: [s][rt(8)][kt(16)][lane*16+idx], flat == tid (2^23 elements)
// ---------------------------------------------------------------------------
__global__ void prep_x(const int* __restrict__ toks, const float* __restrict__ emb,
                       __bf16* __restrict__ dst) {
  int tid = blockIdx.x * 256 + threadIdx.x;
  int idx  = tid & 15;
  int lane = (tid >> 4) & 31;
  int kt   = (tid >> 9) & 15;
  int rt   = (tid >> 13) & 7;
  int s    = tid >> 16;
  int m = lane & 15;
  int k = kt * 32 + afrag_k(lane, idx);
  int b = rt * 16 + m;
  int tok = toks[s * BATCH + b];
  dst[tid] = (__bf16)emb[(size_t)tok * EMB + k];
}

// ---------------------------------------------------------------------------
// Phase 1: persistent recurrence. 8 blocks x 512 threads (16 waves).
// Block rt owns batch rows [rt*16, rt*16+16) for all 128 timesteps.
// h0 = tanh([x , h0p] @ W0^T + b0) ; h1 = tanh([h0 , h1p] @ W1^T + b1)
// Weight fragment loads are re-issued every step (L2-resident) — the base
// pointers are laundered through an asm barrier each iteration so LICM can't
// hoist 4KB/lane of fragments out of the time loop and spill to scratch.
// ---------------------------------------------------------------------------
__global__ __launch_bounds__(512) void rnn_recurrence(
    const __bf16* __restrict__ Xsw, const __bf16* __restrict__ W0sw,
    const __bf16* __restrict__ W1sw, const float* __restrict__ b0,
    const float* __restrict__ b1, const float* __restrict__ hidden0,
    __bf16* __restrict__ H1sw, float* __restrict__ outHidden) {
  __shared__ __bf16 hA0[16 * 512];     // h0_prev / h0_new, A-frag order
  __shared__ __bf16 hA1[16 * 512];     // h1_prev, A-frag order
  __shared__ __bf16 hplain[16 * 512];  // staging, [m][j] row major

  const int tid  = threadIdx.x;
  const int lane = tid & 31;
  const int wave = tid >> 5;            // 0..15 -> ntiles wave*2, wave*2+1
  const int rt   = blockIdx.x;          // 0..7
  const int nt0  = wave * 2, nt1 = wave * 2 + 1;

  // init h state A-fragments from initial hidden [2,B,H] f32
  for (int e = tid; e < 16 * 512; e += 512) {
    int idx = e & 15, ln = (e >> 4) & 31, kt = e >> 9;
    int m = ln & 15;
    int k = kt * 32 + afrag_k(ln, idx);
    hA0[e] = (__bf16)hidden0[(size_t)(rt * 16 + m) * HID + k];
    hA1[e] = (__bf16)hidden0[(size_t)BATCH * HID + (size_t)(rt * 16 + m) * HID + k];
  }
  __syncthreads();

  const int n16 = lane & 15;
  const int mh  = (lane >> 4) * 8;

  const float bz00 = b0[nt0 * 16 + n16];
  const float bz01 = b0[nt1 * 16 + n16];
  const float bz10 = b1[nt0 * 16 + n16];
  const float bz11 = b1[nt1 * 16 + n16];

  unsigned long long w0a = (unsigned long long)W0sw;
  unsigned long long w1a = (unsigned long long)W1sw;

  for (int s = 0; s < SEQ; ++s) {
    // opaque per-iteration redefinition: keeps weight loads inside the loop
    asm volatile("" : "+s"(w0a), "+s"(w1a));
    const __bf16* W0p = (const __bf16*)w0a;
    const __bf16* W1p = (const __bf16*)w1a;
    const __bf16* Xbase = Xsw + (size_t)(s * 8 + rt) * (16 * 512);

    // ---------------- layer 0 ----------------
    v8f acc0, acc1;
    for (int v = 0; v < 8; ++v) { acc0[v] = bz00; acc1[v] = bz01; }
#pragma unroll 4
    for (int kt = 0; kt < 16; ++kt) {  // K 0..511 : x (global, A-frag order)
      v16bf a   = *(const v16bf*)(Xbase + kt * 512 + lane * 16);
      v16bf bw0 = *(const v16bf*)(W0p + (size_t)(nt0 * 32 + kt) * 512 + lane * 16);
      v16bf bw1 = *(const v16bf*)(W0p + (size_t)(nt1 * 32 + kt) * 512 + lane * 16);
      acc0 = WMMA_BF16(a, bw0, acc0);
      acc1 = WMMA_BF16(a, bw1, acc1);
    }
#pragma unroll 4
    for (int kt = 0; kt < 16; ++kt) {  // K 512..1023 : h0_prev (LDS)
      v16bf a   = *(const v16bf*)(hA0 + kt * 512 + lane * 16);
      v16bf bw0 = *(const v16bf*)(W0p + (size_t)(nt0 * 32 + 16 + kt) * 512 + lane * 16);
      v16bf bw1 = *(const v16bf*)(W0p + (size_t)(nt1 * 32 + 16 + kt) * 512 + lane * 16);
      acc0 = WMMA_BF16(a, bw0, acc0);
      acc1 = WMMA_BF16(a, bw1, acc1);
    }
    for (int v = 0; v < 8; ++v) {        // tanh -> plain staging (+ final dump)
      int m = mh + v;
      float t0 = tanhf(acc0[v]);
      float t1 = tanhf(acc1[v]);
      hplain[m * 512 + nt0 * 16 + n16] = (__bf16)t0;
      hplain[m * 512 + nt1 * 16 + n16] = (__bf16)t1;
      if (s == SEQ - 1) {
        outHidden[(size_t)(rt * 16 + m) * HID + nt0 * 16 + n16] = t0;
        outHidden[(size_t)(rt * 16 + m) * HID + nt1 * 16 + n16] = t1;
      }
    }
    __syncthreads();
    for (int e = tid; e < 16 * 512; e += 512) {   // rebuild hA0 (A-frag of h0_new)
      int idx = e & 15, ln = (e >> 4) & 31, kt = e >> 9;
      int m = ln & 15;
      int k = kt * 32 + afrag_k(ln, idx);
      hA0[e] = hplain[m * 512 + k];
    }
    __syncthreads();

    // ---------------- layer 1 ----------------
    for (int v = 0; v < 8; ++v) { acc0[v] = bz10; acc1[v] = bz11; }
#pragma unroll 4
    for (int kt = 0; kt < 16; ++kt) {  // K 0..511 : h0_new (LDS)
      v16bf a   = *(const v16bf*)(hA0 + kt * 512 + lane * 16);
      v16bf bw0 = *(const v16bf*)(W1p + (size_t)(nt0 * 32 + kt) * 512 + lane * 16);
      v16bf bw1 = *(const v16bf*)(W1p + (size_t)(nt1 * 32 + kt) * 512 + lane * 16);
      acc0 = WMMA_BF16(a, bw0, acc0);
      acc1 = WMMA_BF16(a, bw1, acc1);
    }
#pragma unroll 4
    for (int kt = 0; kt < 16; ++kt) {  // K 512..1023 : h1_prev (LDS)
      v16bf a   = *(const v16bf*)(hA1 + kt * 512 + lane * 16);
      v16bf bw0 = *(const v16bf*)(W1p + (size_t)(nt0 * 32 + 16 + kt) * 512 + lane * 16);
      v16bf bw1 = *(const v16bf*)(W1p + (size_t)(nt1 * 32 + 16 + kt) * 512 + lane * 16);
      acc0 = WMMA_BF16(a, bw0, acc0);
      acc1 = WMMA_BF16(a, bw1, acc1);
    }
    for (int v = 0; v < 8; ++v) {
      int m = mh + v;
      float t0 = tanhf(acc0[v]);
      float t1 = tanhf(acc1[v]);
      hplain[m * 512 + nt0 * 16 + n16] = (__bf16)t0;
      hplain[m * 512 + nt1 * 16 + n16] = (__bf16)t1;
      if (s == SEQ - 1) {
        size_t l1 = (size_t)BATCH * HID;
        outHidden[l1 + (size_t)(rt * 16 + m) * HID + nt0 * 16 + n16] = t0;
        outHidden[l1 + (size_t)(rt * 16 + m) * HID + nt1 * 16 + n16] = t1;
      }
    }
    __syncthreads();
    {  // rebuild hA1 + stream h1 A-fragments to global for phase 2
      __bf16* H1base = H1sw + (size_t)(s * 8 + rt) * (16 * 512);
      for (int e = tid; e < 16 * 512; e += 512) {
        int idx = e & 15, ln = (e >> 4) & 31, kt = e >> 9;
        int m = ln & 15;
        int k = kt * 32 + afrag_k(ln, idx);
        __bf16 vv = hplain[m * 512 + k];
        hA1[e] = vv;
        H1base[e] = vv;
      }
    }
    __syncthreads();
  }
}

// ---------------------------------------------------------------------------
// Phase 2: logits[16384, 10000] = H1[16384,512] @ Wout^T + bout
// 1024 row-tiles x 625 col-tiles; each wave does 1 row-tile x 5 col-tiles.
// 128000 waves -> 16000 blocks of 256 threads. Non-temporal logits stores.
// ---------------------------------------------------------------------------
__global__ __launch_bounds__(256) void rnn_logits(
    const __bf16* __restrict__ H1sw, const __bf16* __restrict__ Woutsw,
    const float* __restrict__ bout, float* __restrict__ logits) {
  const int lane  = threadIdx.x & 31;
  const int wtask = blockIdx.x * 8 + (threadIdx.x >> 5);  // 0..127999
  const int rt  = wtask / 125;          // row tile 0..1023
  const int ct0 = (wtask % 125) * 5;    // first col tile (of 625)

  const int n16 = lane & 15;
  v8f acc[5];
#pragma unroll
  for (int i = 0; i < 5; ++i) {
    float bz = bout[(ct0 + i) * 16 + n16];
    for (int v = 0; v < 8; ++v) acc[i][v] = bz;
  }

  const __bf16* Abase = H1sw + (size_t)rt * (16 * 512);
#pragma unroll 4
  for (int kt = 0; kt < 16; ++kt) {
    v16bf a = *(const v16bf*)(Abase + kt * 512 + lane * 16);
#pragma unroll
    for (int i = 0; i < 5; ++i) {
      v16bf b = *(const v16bf*)(Woutsw + ((size_t)((ct0 + i) * 16 + kt)) * 512 + lane * 16);
      acc[i] = WMMA_BF16(a, b, acc[i]);
    }
  }

  const int mh = (lane >> 4) * 8;
#pragma unroll
  for (int i = 0; i < 5; ++i)
    for (int v = 0; v < 8; ++v) {
      int row = rt * 16 + mh + v;
      int col = (ct0 + i) * 16 + n16;
      __builtin_nontemporal_store(acc[i][v], &logits[(size_t)row * VOCAB + col]);
    }
}

// ---------------------------------------------------------------------------
extern "C" void kernel_launch(void* const* d_in, const int* in_sizes, int n_in,
                              void* d_out, int out_size, void* d_ws, size_t ws_size,
                              hipStream_t stream) {
  const int*   toks = (const int*)d_in[0];
  const float* hid0 = (const float*)d_in[1];
  const float* emb  = (const float*)d_in[2];
  const float* W0   = (const float*)d_in[3];
  const float* b0   = (const float*)d_in[4];
  const float* W1   = (const float*)d_in[5];
  const float* b1   = (const float*)d_in[6];
  const float* Wout = (const float*)d_in[7];
  const float* bout = (const float*)d_in[8];
  float* out = (float*)d_out;

  // workspace carve-up (all region sizes multiples of 32B)
  __bf16* W0sw   = (__bf16*)d_ws;                        // 512*1024
  __bf16* W1sw   = W0sw + 512 * 1024;                    // 512*1024
  __bf16* Woutsw = W1sw + 512 * 1024;                    // 10000*512
  __bf16* Xsw    = Woutsw + (size_t)VOCAB * HID;         // 2^23
  __bf16* H1sw   = Xsw + (size_t)SEQ * BATCH * EMB;      // 2^23

  // weight swizzles (bf16 B-fragment order)
  prep_w<<<(512 * 1024) / 256, 256, 0, stream>>>(W0, W0sw, 512 * 1024, 5, 1024);
  prep_w<<<(512 * 1024) / 256, 256, 0, stream>>>(W1, W1sw, 512 * 1024, 5, 1024);
  prep_w<<<(VOCAB * HID) / 256, 256, 0, stream>>>(Wout, Woutsw, VOCAB * HID, 4, 512);
  // embedding gather -> A-fragment order
  prep_x<<<(SEQ * BATCH * EMB) / 256, 256, 0, stream>>>(toks, emb, Xsw);

  // sequential recurrence (8 WGPs, batch-parallel), hidden_final -> tail of out
  float* outHidden = out + (size_t)SEQ * BATCH * VOCAB;
  rnn_recurrence<<<8, 512, 0, stream>>>(Xsw, W0sw, W1sw, b0, b1, hid0, H1sw, outHidden);

  // big output-projection GEMM
  rnn_logits<<<16000, 256, 0, stream>>>(H1sw, Woutsw, bout, out);
}
